// CausalSelfAttention_56650618634861
// MI455X (gfx1250) — compile-verified, matrix operand order checked by reading, not run on hardware
//
#include <hip/hip_runtime.h>

typedef __bf16 bf16_t;
typedef __attribute__((ext_vector_type(16))) __bf16 v16bf;
typedef __attribute__((ext_vector_type(8)))  __bf16 v8bf;
typedef __attribute__((ext_vector_type(4)))  __bf16 v4bf;
typedef __attribute__((ext_vector_type(8)))  float  v8f;
typedef __attribute__((ext_vector_type(4)))  float  v4f;

constexpr int BB  = 4;
constexpr int SS  = 2048;
constexpr int DD  = 768;
constexpr int HH  = 12;
constexpr int HD  = 64;
constexpr int ND3 = 3 * DD;  // 2304

// Hardware bf16 convert (RTNE on gfx1250).
__device__ __forceinline__ bf16_t f2bf(float f) { return (bf16_t)f; }

// Symmetric 16x32 bf16 A/B fragment loader (ISA 7.12.2 layout):
// lanes 0-15: row = row0+lane, K elems {0..7, 16..23}
// lanes 16-31: row = row0+lane-16, K elems {8..15, 24..31}
__device__ __forceinline__ v16bf load_frag_bf16(const bf16_t* base, int lda,
                                                int row0, int k0, int lane) {
  const bf16_t* p = base + (size_t)(row0 + (lane & 15)) * lda + k0 + ((lane >> 4) << 3);
  v8bf lo = *(const v8bf*)p;
  v8bf hi = *(const v8bf*)(p + 16);
  v16bf a;
#pragma unroll
  for (int i = 0; i < 8; ++i) { a[i] = lo[i]; a[i + 8] = hi[i]; }
  return a;
}

// ---------------- elementwise fp32 -> bf16 (x pre-conversion) ----------------
__global__ __launch_bounds__(256) void cvt_f32_bf16_kernel(
    const float* __restrict__ in, bf16_t* __restrict__ out, size_t n) {
  size_t idx = ((size_t)blockIdx.x * 256 + threadIdx.x) * 4;
  if (idx >= n) return;
  v4f x = *(const v4f*)(in + idx);
  v4bf y;
#pragma unroll
  for (int i = 0; i < 4; ++i) y[i] = f2bf(x[i]);
  *(v4bf*)(out + idx) = y;
}

// ---------------- weight prep: fp32 [K,N] -> bf16 [N,K] ----------------
__global__ __launch_bounds__(256) void transpose_cvt_kernel(
    const float* __restrict__ in, bf16_t* __restrict__ out, int K, int N) {
  size_t idx = (size_t)blockIdx.x * 256 + threadIdx.x;
  if (idx >= (size_t)K * N) return;
  int n = (int)(idx / K);
  int k = (int)(idx % K);
  out[idx] = f2bf(in[(size_t)k * N + n]);
}

// Ping-pong double-buffered WMMA mainloop for C[16x64] += A[16xK] * B[Kx64].
// No register rotation: buffers are refilled in place, loads of one phase fly
// while the other phase's 4 WMMAs execute.
__device__ __forceinline__ void gemm_mainloop(const bf16_t* __restrict__ A, int mt,
                                              const bf16_t* __restrict__ wT, int nt,
                                              int lane, v8f acc[4]) {
  v16bf a0 = load_frag_bf16(A, DD, mt, 0, lane);
  v16bf b0[4];
#pragma unroll
  for (int t = 0; t < 4; ++t) b0[t] = load_frag_bf16(wT, DD, nt + 16 * t, 0, lane);

  for (int k0 = 0; k0 < DD; k0 += 64) {
    // phase 1 loads (k0+32) while phase 0 WMMAs run
    v16bf a1 = load_frag_bf16(A, DD, mt, k0 + 32, lane);
    v16bf b1[4];
#pragma unroll
    for (int t = 0; t < 4; ++t) b1[t] = load_frag_bf16(wT, DD, nt + 16 * t, k0 + 32, lane);
#pragma unroll
    for (int t = 0; t < 4; ++t)
      acc[t] = __builtin_amdgcn_wmma_f32_16x16x32_bf16(false, a0, false, b0[t],
                                                       (short)0, acc[t], false, false);
    // phase 0 reload (k0+64) while phase 1 WMMAs run
    if (k0 + 64 < DD) {
      a0 = load_frag_bf16(A, DD, mt, k0 + 64, lane);
#pragma unroll
      for (int t = 0; t < 4; ++t) b0[t] = load_frag_bf16(wT, DD, nt + 16 * t, k0 + 64, lane);
    }
#pragma unroll
    for (int t = 0; t < 4; ++t)
      acc[t] = __builtin_amdgcn_wmma_f32_16x16x32_bf16(false, a1, false, b1[t],
                                                       (short)0, acc[t], false, false);
  }
}

// ---------------- QKV projection: xbf[8192,768] @ wT[2304,768] + bias ----------------
// Wave tile: 16(M) x 64(N). Epilogue routes to Q/K [b,s,h*64+hd] bf16 and V
// transposed [b,h,hd,s] bf16 so attention B-fragments are contiguous loads.
__global__ __launch_bounds__(256) void qkv_gemm_kernel(
    const bf16_t* __restrict__ xbf, const bf16_t* __restrict__ wT,
    const float* __restrict__ bias, bf16_t* __restrict__ qbuf,
    bf16_t* __restrict__ kbuf, bf16_t* __restrict__ vT) {
  const int lane = threadIdx.x & 31;
  const int wv   = threadIdx.x >> 5;
  const int mt = blockIdx.y * 128 + wv * 16;
  const int nt = blockIdx.x * 64;
  const int nl = lane & 15, half = lane >> 4;

  v8f acc[4];
#pragma unroll
  for (int t = 0; t < 4; ++t) {
    float bv = bias[nt + 16 * t + nl];
#pragma unroll
    for (int i = 0; i < 8; ++i) acc[t][i] = bv;
  }

  gemm_mainloop(xbf, mt, wT, nt, lane, acc);

#pragma unroll
  for (int t = 0; t < 4; ++t) {
    int col = nt + 16 * t + nl;
#pragma unroll
    for (int i = 0; i < 8; ++i) {
      int row = mt + i + 8 * half;  // row = b*SS + s
      float v = acc[t][i];
      if (col < DD) {
        qbuf[(size_t)row * DD + col] = f2bf(v);
      } else if (col < 2 * DD) {
        kbuf[(size_t)row * DD + (col - DD)] = f2bf(v);
      } else {
        int cv = col - 2 * DD;
        int h = cv >> 6, hd = cv & 63;
        int bb = row >> 11, s = row & (SS - 1);
        vT[(((size_t)bb * HH + h) * HD + hd) * SS + s] = f2bf(v);
      }
    }
  }
}

// ---------------- causal flash attention ----------------
// One wave = one 16-query tile x full head dim (64). 32-key blocks, online softmax.
__global__ __launch_bounds__(256) void attn_kernel(
    const bf16_t* __restrict__ qbuf, const bf16_t* __restrict__ kbuf,
    const bf16_t* __restrict__ vT, bf16_t* __restrict__ ybuf) {
  __shared__ __align__(16) bf16_t plds[8][16][32];  // per-wave P tile (bf16)
  const int lane = threadIdx.x & 31;
  const int wv   = threadIdx.x >> 5;
  const int b = blockIdx.z, h = blockIdx.y;
  const int qt = blockIdx.x * 128 + wv * 16;
  const int nl = lane & 15, half = lane >> 4;

  const bf16_t* Qb = qbuf + (size_t)b * SS * DD + h * HD;  // row s, lda DD
  const bf16_t* Kb = kbuf + (size_t)b * SS * DD + h * HD;
  const bf16_t* Vb = vT + ((size_t)b * HH + h) * HD * SS;  // row hd, lda SS

  v16bf qa0 = load_frag_bf16(Qb, DD, qt, 0, lane);
  v16bf qa1 = load_frag_bf16(Qb, DD, qt, 32, lane);

  v8f o[4];
  float m_i[8], l_i[8];
#pragma unroll
  for (int t = 0; t < 4; ++t)
#pragma unroll
    for (int i = 0; i < 8; ++i) o[t][i] = 0.f;
#pragma unroll
  for (int i = 0; i < 8; ++i) { m_i[i] = -1e30f; l_i[i] = 0.f; }

  const float scale = 0.125f;  // 1/sqrt(64)
  const int qmax = qt + 15;

  for (int kt = 0; kt <= qmax; kt += 32) {
    const bool tile1 = (kt + 16) <= qmax;  // wave-uniform: is the 2nd key sub-tile live?
    if (kt + 32 <= qmax)  // prefetch next key block of K (streamed operand)
      __builtin_prefetch(Kb + (size_t)(kt + 32 + nl) * DD, 0, 1);

    v8f s0, s1;
#pragma unroll
    for (int i = 0; i < 8; ++i) { s0[i] = 0.f; s1[i] = 0.f; }
    {
      v16bf kb00 = load_frag_bf16(Kb, DD, kt, 0, lane);
      v16bf kb01 = load_frag_bf16(Kb, DD, kt, 32, lane);
      s0 = __builtin_amdgcn_wmma_f32_16x16x32_bf16(false, qa0, false, kb00, (short)0, s0, false, false);
      s0 = __builtin_amdgcn_wmma_f32_16x16x32_bf16(false, qa1, false, kb01, (short)0, s0, false, false);
    }
    if (tile1) {
      v16bf kb10 = load_frag_bf16(Kb, DD, kt + 16, 0, lane);
      v16bf kb11 = load_frag_bf16(Kb, DD, kt + 16, 32, lane);
      s1 = __builtin_amdgcn_wmma_f32_16x16x32_bf16(false, qa0, false, kb10, (short)0, s1, false, false);
      s1 = __builtin_amdgcn_wmma_f32_16x16x32_bf16(false, qa1, false, kb11, (short)0, s1, false, false);
    }

    float rescale[8];
#pragma unroll
    for (int i = 0; i < 8; ++i) {
      const int qrow = qt + i + 8 * half;
      float a0 = s0[i] * scale;
      float a1 = tile1 ? s1[i] * scale : -1e30f;
      if (kt + nl > qrow) a0 = -1e30f;
      if (kt + 16 + nl > qrow) a1 = -1e30f;
      float mx = fmaxf(a0, a1);
#pragma unroll
      for (int off = 1; off < 16; off <<= 1) mx = fmaxf(mx, __shfl_xor(mx, off, 16));
      const float mnew = fmaxf(m_i[i], mx);
      const float r  = __expf(m_i[i] - mnew);
      const float p0 = __expf(a0 - mnew);
      const float p1 = __expf(a1 - mnew);
      float rs = p0 + p1;
#pragma unroll
      for (int off = 1; off < 16; off <<= 1) rs += __shfl_xor(rs, off, 16);
      m_i[i] = mnew;
      l_i[i] = l_i[i] * r + rs;
      rescale[i] = r;
      plds[wv][i + 8 * half][nl]      = f2bf(p0);
      plds[wv][i + 8 * half][nl + 16] = f2bf(p1);
    }
#pragma unroll
    for (int t = 0; t < 4; ++t)
#pragma unroll
      for (int i = 0; i < 8; ++i) o[t][i] *= rescale[i];

    asm volatile("s_wait_dscnt 0" ::: "memory");  // P tile visible to all lanes of this wave

    v16bf pa = load_frag_bf16(&plds[wv][0][0], 32, 0, 0, lane);
#pragma unroll
    for (int t = 0; t < 4; ++t) {
      v16bf vb = load_frag_bf16(Vb, SS, 16 * t, kt, lane);
      o[t] = __builtin_amdgcn_wmma_f32_16x16x32_bf16(false, pa, false, vb, (short)0, o[t], false, false);
    }
  }

#pragma unroll
  for (int t = 0; t < 4; ++t)
#pragma unroll
    for (int i = 0; i < 8; ++i) {
      int row = qt + i + 8 * half;
      float v = o[t][i] / l_i[i];
      ybuf[((size_t)b * SS + row) * DD + h * HD + 16 * t + nl] = f2bf(v);
    }
}

// ---------------- output projection: y[8192,768] @ wT[768,768] + bias -> fp32 ----------------
__global__ __launch_bounds__(256) void proj_gemm_kernel(
    const bf16_t* __restrict__ y, const bf16_t* __restrict__ wT,
    const float* __restrict__ bias, float* __restrict__ out) {
  const int lane = threadIdx.x & 31;
  const int wv   = threadIdx.x >> 5;
  const int mt = blockIdx.y * 128 + wv * 16;
  const int nt = blockIdx.x * 64;
  const int nl = lane & 15, half = lane >> 4;

  v8f acc[4];
#pragma unroll
  for (int t = 0; t < 4; ++t) {
    float bv = bias[nt + 16 * t + nl];
#pragma unroll
    for (int i = 0; i < 8; ++i) acc[t][i] = bv;
  }

  gemm_mainloop(y, mt, wT, nt, lane, acc);

#pragma unroll
  for (int t = 0; t < 4; ++t)
#pragma unroll
    for (int i = 0; i < 8; ++i)
      out[(size_t)(mt + i + 8 * half) * DD + nt + 16 * t + nl] = acc[t][i];
}

extern "C" void kernel_launch(void* const* d_in, const int* in_sizes, int n_in,
                              void* d_out, int out_size, void* d_ws, size_t ws_size,
                              hipStream_t stream) {
  (void)in_sizes; (void)n_in; (void)out_size; (void)ws_size;
  const float* x      = (const float*)d_in[0];
  const float* w_qkv  = (const float*)d_in[1];
  const float* b_qkv  = (const float*)d_in[2];
  const float* w_proj = (const float*)d_in[3];
  const float* b_proj = (const float*)d_in[4];
  float* out = (float*)d_out;

  char* ws = (char*)d_ws;
  size_t off = 0;
  bf16_t* xbf    = (bf16_t*)(ws + off); off += (size_t)BB * SS * DD * sizeof(bf16_t);
  bf16_t* wqkvT  = (bf16_t*)(ws + off); off += (size_t)ND3 * DD * sizeof(bf16_t);
  bf16_t* wprojT = (bf16_t*)(ws + off); off += (size_t)DD * DD * sizeof(bf16_t);
  bf16_t* qbuf   = (bf16_t*)(ws + off); off += (size_t)BB * SS * DD * sizeof(bf16_t);
  bf16_t* kbuf   = (bf16_t*)(ws + off); off += (size_t)BB * SS * DD * sizeof(bf16_t);
  bf16_t* vT     = (bf16_t*)(ws + off); off += (size_t)BB * SS * DD * sizeof(bf16_t);
  bf16_t* ybuf   = (bf16_t*)(ws + off); off += (size_t)BB * SS * DD * sizeof(bf16_t);

  {
    size_t n = (size_t)BB * SS * DD;  // 6,291,456 (divisible by 1024)
    cvt_f32_bf16_kernel<<<(unsigned)(n / 1024), 256, 0, stream>>>(x, xbf, n);
  }
  {
    size_t tot = (size_t)DD * ND3;
    transpose_cvt_kernel<<<(unsigned)((tot + 255) / 256), 256, 0, stream>>>(w_qkv, wqkvT, DD, ND3);
  }
  {
    size_t tot = (size_t)DD * DD;
    transpose_cvt_kernel<<<(unsigned)((tot + 255) / 256), 256, 0, stream>>>(w_proj, wprojT, DD, DD);
  }
  qkv_gemm_kernel<<<dim3(ND3 / 64, (BB * SS) / 128), 256, 0, stream>>>(xbf, wqkvT, b_qkv, qbuf, kbuf, vT);
  attn_kernel<<<dim3(SS / 128, HH, BB), 256, 0, stream>>>(qbuf, kbuf, vT, ybuf);
  proj_gemm_kernel<<<dim3(DD / 64, (BB * SS) / 128), 256, 0, stream>>>(ybuf, wprojT, b_proj, out);
}
